// DifferentiablePriorityBuffer_11192684773814
// MI455X (gfx1250) — compile-verified, matrix operand order checked by reading, not run on hardware
//
#include <hip/hip_runtime.h>

// Problem constants (match reference)
#define Bc 64
#define Tc 512
#define Dc 2048
#define Nc 16384
#define ROUNDSc 10
#define KSPLIT 8   // K-split for aw @ values GEMM

typedef float  v8f    __attribute__((ext_vector_type(8)));
typedef float  v2f    __attribute__((ext_vector_type(2)));
typedef __bf16 v16bf  __attribute__((ext_vector_type(16)));
typedef __bf16 bf16x8 __attribute__((ext_vector_type(8)));

// ---------------- workspace layout (bytes) ----------------
// scalars: [0]=stopped, [1]=wsum, [2]=inc (this round), [3]=inc*rw (this round)
static constexpr size_t OFF_SCAL  = 0;
static constexpr size_t OFF_WAGES = 256;
static constexpr size_t OFF_EFF   = OFF_WAGES + (size_t)Nc * 4;
static constexpr size_t OFF_POOL  = OFF_EFF   + (size_t)Nc * 4;
static constexpr size_t OFF_QUERY = OFF_POOL  + (size_t)Bc * Dc * 4;
static constexpr size_t OFF_BS    = OFF_QUERY + (size_t)Bc * Dc * 4;
static constexpr size_t OFF_AWF   = OFF_BS    + (size_t)Bc * Nc * 4;
static constexpr size_t OFF_ACC   = OFF_AWF   + (size_t)Bc * Nc * 4;
static constexpr size_t OFF_PART  = OFF_ACC   + (size_t)Bc * Dc * 4;
static constexpr size_t OFF_AWB   = OFF_PART  + (size_t)KSPLIT * Bc * Dc * 4;
static constexpr size_t OFF_VT    = OFF_AWB   + (size_t)Bc * Nc * 2;
// OFF_VT consumes Dc*Nc*2 = 64MB (values transposed, bf16). Total ~80MB.

// ---------------- kernels ----------------

__global__ void k_init(const float* __restrict__ ages, float* __restrict__ wages,
                       float* __restrict__ acc, float* __restrict__ scalars) {
  int i = blockIdx.x * blockDim.x + threadIdx.x;
  if (i < Nc) wages[i] = ages[i];
  if (i < Bc * Dc) acc[i] = 0.f;
  if (i == 0) { scalars[0] = 0.f; scalars[1] = 0.f; scalars[2] = 0.f; scalars[3] = 0.f; }
}

// values[N][D] f32  ->  vt[D][N] bf16  (LDS-tiled 32x32 transpose, coalesced both ways)
__global__ void k_cvt_transpose(const float* __restrict__ v, __bf16* __restrict__ vt) {
  __shared__ __bf16 tile[32][33];
  int tx = threadIdx.x, ty = threadIdx.y;           // 32 x 8
  int n0 = blockIdx.x * 32, d0 = blockIdx.y * 32;
#pragma unroll
  for (int i = 0; i < 4; ++i) {
    int r = ty + 8 * i;
    tile[r][tx] = (__bf16)v[(size_t)(n0 + r) * Dc + d0 + tx];
  }
  __syncthreads();
#pragma unroll
  for (int i = 0; i < 4; ++i) {
    int r = ty + 8 * i;
    vt[(size_t)(d0 + r) * Nc + n0 + tx] = tile[tx][r];
  }
}

// masked mean-pool over T: one block of 256 threads handles 256 consecutive d of one b
__global__ void k_pool(const float* __restrict__ qs, const float* __restrict__ mask,
                       float* __restrict__ pooled) {
  __shared__ float mrow[Tc];
  int idx0 = blockIdx.x * 256;
  int b = idx0 / Dc;
  int d = (idx0 % Dc) + threadIdx.x;
  for (int t = threadIdx.x; t < Tc; t += 256) mrow[t] = mask[b * Tc + t];
  __syncthreads();
  float msum = 0.f, s = 0.f;
  for (int t = 0; t < Tc; ++t) {
    float m = mrow[t];
    msum += m;
    s += qs[((size_t)(b * Tc + t)) * Dc + d] * m;
  }
  pooled[(size_t)b * Dc + d] = s / (msum + 1e-8f);
}

// C[64 x ncols] = scale * (A[64 x K] @ W[ncols x K]^T) + bias   (f32 WMMA 16x16x4)
// flags&1: multiply by 1/max(wsum,1e-8) and zero output if wsum<=0
__global__ void k_gemm_f32(const float* __restrict__ A, const float* __restrict__ W,
                           const float* __restrict__ bias, float* __restrict__ C,
                           int ncols, int K, float scale,
                           const float* __restrict__ scalars, int flags) {
  int wid  = (blockIdx.x * blockDim.x + threadIdx.x) >> 5;
  int lane = threadIdx.x & 31;
  int ntiles = ncols >> 4;
  if (wid >= 4 * ntiles) return;
  int mbase = (wid & 3) * 16;
  int nbase = (wid >> 2) * 16;
  int l15 = lane & 15, h = lane >> 4;

  const float* ap = A + (size_t)(mbase + l15) * K + 2 * h;
  const float* bp = W + (size_t)(nbase + l15) * K + 2 * h;

  v8f c = {};
  for (int k = 0; k < K; k += 4) {
    v2f a = *(const v2f*)(ap + k);
    v2f b = *(const v2f*)(bp + k);
    c = __builtin_amdgcn_wmma_f32_16x16x4_f32(false, a, false, b, (short)0, c,
                                              false, false);
  }

  float mul = scale;
  bool zero = false;
  if (flags & 1) {
    float ws = scalars[1];
    if (ws > 0.f) mul = scale / fmaxf(ws, 1e-8f); else zero = true;
  }
  float bcol = bias ? bias[nbase + l15] : 0.f;
#pragma unroll
  for (int v = 0; v < 8; ++v) {
    int row = mbase + v + 8 * h;
    float val = zero ? 0.f : (c[v] * mul + bcol);
    C[(size_t)row * ncols + nbase + l15] = val;
  }
}

// per-round gate: eff, active, stopped/inc/wsum bookkeeping (single block)
__global__ void k_round_gate(const float* __restrict__ prio,
                             const unsigned char* __restrict__ valid,
                             const float* __restrict__ wages,
                             float* __restrict__ effact,
                             float* __restrict__ scalars, float rw) {
  __shared__ float red[1024];
  float s = 0.f;
  for (int n = threadIdx.x; n < Nc; n += 1024) {
    float age = wages[n];
    float eff = prio[n] * expf(age * -0.1053605157f);     // DECAY^age
    float act = valid[n] ? (1.f / (1.f + expf(-10.f * (eff - 0.5f)))) : 0.f;
    s += act;
    effact[n] = act * eff;
  }
  red[threadIdx.x] = s;
  __syncthreads();
  for (int o = 512; o > 0; o >>= 1) {
    if ((int)threadIdx.x < o) red[threadIdx.x] += red[threadIdx.x + o];
    __syncthreads();
  }
  if (threadIdx.x == 0) {
    float stp = scalars[0];
    if (red[0] < 0.5f) stp = 1.f;
    scalars[0] = stp;
    float inc = (stp > 0.f) ? 0.f : 1.f;
    scalars[2] = inc;
    scalars[3] = inc * rw;
    scalars[1] += inc * rw;
  }
}

// row softmax over N, writes f32 (for column-mean) and bf16 (for WMMA A)
__global__ void k_softmax(const float* __restrict__ bscores,
                          const float* __restrict__ effact,
                          const unsigned char* __restrict__ valid,
                          float* __restrict__ awf, __bf16* __restrict__ awb) {
  __shared__ float red[256];
  int b = blockIdx.x, tid = threadIdx.x;
  const float* row = bscores + (size_t)b * Nc;
  float m = -__builtin_inff();
  for (int n = tid; n < Nc; n += 256) {
    float sc = valid[n] ? row[n] * effact[n] : -__builtin_inff();
    m = fmaxf(m, sc);
  }
  red[tid] = m; __syncthreads();
  for (int o = 128; o > 0; o >>= 1) {
    if (tid < o) red[tid] = fmaxf(red[tid], red[tid + o]);
    __syncthreads();
  }
  float rowmax = red[0];
  __syncthreads();
  float s = 0.f;
  for (int n = tid; n < Nc; n += 256) {
    float sc = valid[n] ? row[n] * effact[n] : -__builtin_inff();
    s += expf(sc - rowmax);
  }
  red[tid] = s; __syncthreads();
  for (int o = 128; o > 0; o >>= 1) {
    if (tid < o) red[tid] += red[tid + o];
    __syncthreads();
  }
  float inv = 1.f / red[0];
  for (int n = tid; n < Nc; n += 256) {
    float sc = valid[n] ? row[n] * effact[n] : -__builtin_inff();
    float e = expf(sc - rowmax) * inv;
    awf[(size_t)b * Nc + n] = e;
    awb[(size_t)b * Nc + n] = (__bf16)e;
  }
}

__device__ __forceinline__ void ld8(v16bf& a, int off, const __bf16* p) {
  bf16x8 t = *(const bf16x8*)p;   // one 16-byte global_load_b128
#pragma unroll
  for (int i = 0; i < 8; ++i) a[off + i] = t[i];
}

// part[kc][64 x 2048] = aw_bf16[64 x 16384] @ values^T_bf16 (K-chunk kc)
// wave = one 16-wide D strip x full M=64, 8-way K split. bf16 WMMA, f32 accum.
// Both A and B fragments are two contiguous b128 loads per lane (vt is [D][N]).
__global__ void k_gemm_out(const __bf16* __restrict__ awb,
                           const __bf16* __restrict__ vt,
                           float* __restrict__ part) {
  int wid  = (blockIdx.x * blockDim.x + threadIdx.x) >> 5;  // 0..1023
  int lane = threadIdx.x & 31;
  int strip = wid & ((Dc / 16) - 1);   // 0..127
  int kc    = wid >> 7;                // 0..7
  int l15 = lane & 15, h = lane >> 4;
  int col = strip * 16 + l15;
  const int k0 = kc * (Nc / KSPLIT), k1 = k0 + (Nc / KSPLIT);

  v8f c0 = {}, c1 = {}, c2 = {}, c3 = {};
  const __bf16* a0 = awb + (size_t)(0  + l15) * Nc;
  const __bf16* a1 = awb + (size_t)(16 + l15) * Nc;
  const __bf16* a2 = awb + (size_t)(32 + l15) * Nc;
  const __bf16* a3 = awb + (size_t)(48 + l15) * Nc;
  const __bf16* bp = vt  + (size_t)col * Nc;     // column `col` of values = row of vt

  for (int k = k0; k < k1; k += 32) {
    v16bf bf;
    ld8(bf, 0, bp + k + 8 * h); ld8(bf, 8, bp + k + 16 + 8 * h);

    v16bf af;
    ld8(af, 0, a0 + k + 8 * h); ld8(af, 8, a0 + k + 16 + 8 * h);
    c0 = __builtin_amdgcn_wmma_f32_16x16x32_bf16(false, af, false, bf, (short)0, c0, false, false);
    ld8(af, 0, a1 + k + 8 * h); ld8(af, 8, a1 + k + 16 + 8 * h);
    c1 = __builtin_amdgcn_wmma_f32_16x16x32_bf16(false, af, false, bf, (short)0, c1, false, false);
    ld8(af, 0, a2 + k + 8 * h); ld8(af, 8, a2 + k + 16 + 8 * h);
    c2 = __builtin_amdgcn_wmma_f32_16x16x32_bf16(false, af, false, bf, (short)0, c2, false, false);
    ld8(af, 0, a3 + k + 8 * h); ld8(af, 8, a3 + k + 16 + 8 * h);
    c3 = __builtin_amdgcn_wmma_f32_16x16x32_bf16(false, af, false, bf, (short)0, c3, false, false);
  }

  float* p = part + (size_t)kc * Bc * Dc;
#pragma unroll
  for (int v = 0; v < 8; ++v) {
    p[(size_t)(0  + v + 8 * h) * Dc + col] = c0[v];
    p[(size_t)(16 + v + 8 * h) * Dc + col] = c1[v];
    p[(size_t)(32 + v + 8 * h) * Dc + col] = c2[v];
    p[(size_t)(48 + v + 8 * h) * Dc + col] = c3[v];
  }
}

__global__ void k_reduce_part(const float* __restrict__ part, float* __restrict__ acc,
                              const float* __restrict__ scalars) {
  int i = blockIdx.x * blockDim.x + threadIdx.x;  // < Bc*Dc
  float s = 0.f;
#pragma unroll
  for (int kc = 0; kc < KSPLIT; ++kc) s += part[(size_t)kc * Bc * Dc + i];
  acc[i] += scalars[3] * s;
}

__global__ void k_colmean(const float* __restrict__ awf, float* __restrict__ wages,
                          const float* __restrict__ scalars) {
  int n = blockIdx.x * blockDim.x + threadIdx.x;  // < Nc
  float s = 0.f;
  for (int b = 0; b < Bc; ++b) s += awf[(size_t)b * Nc + n];
  wages[n] += scalars[2] * (s * (1.f / (float)Bc));
}

// ---------------- host ----------------

extern "C" void kernel_launch(void* const* d_in, const int* in_sizes, int n_in,
                              void* d_out, int out_size, void* d_ws, size_t ws_size,
                              hipStream_t stream) {
  (void)in_sizes; (void)n_in; (void)out_size; (void)ws_size;
  const float*         qs    = (const float*)d_in[0];
  const float*         amask = (const float*)d_in[1];
  const float*         keys  = (const float*)d_in[2];
  const float*         vals  = (const float*)d_in[3];
  const float*         prio  = (const float*)d_in[4];
  const float*         ages  = (const float*)d_in[5];
  const unsigned char* valid = (const unsigned char*)d_in[6];
  const float*         Wq    = (const float*)d_in[7];
  const float*         bq    = (const float*)d_in[8];
  const float*         Wc    = (const float*)d_in[9];
  const float*         bc    = (const float*)d_in[10];
  float* out = (float*)d_out;

  char* w = (char*)d_ws;
  float*  scalars = (float*)(w + OFF_SCAL);
  float*  wages   = (float*)(w + OFF_WAGES);
  float*  effact  = (float*)(w + OFF_EFF);
  float*  pooled  = (float*)(w + OFF_POOL);
  float*  query   = (float*)(w + OFF_QUERY);
  float*  bscores = (float*)(w + OFF_BS);
  float*  awf     = (float*)(w + OFF_AWF);
  float*  acc     = (float*)(w + OFF_ACC);
  float*  part    = (float*)(w + OFF_PART);
  __bf16* awb     = (__bf16*)(w + OFF_AWB);
  __bf16* vt      = (__bf16*)(w + OFF_VT);

  // init + values->bf16 transpose + pool
  k_init<<<512, 256, 0, stream>>>(ages, wages, acc, scalars);
  {
    dim3 grid(Nc / 32, Dc / 32), blk(32, 8);
    k_cvt_transpose<<<grid, blk, 0, stream>>>(vals, vt);
  }
  k_pool<<<(Bc * Dc) / 256, 256, 0, stream>>>(qs, amask, pooled);

  // query = pooled @ Wq^T + bq   (f32 WMMA)
  k_gemm_f32<<<Dc / 16, 128, 0, stream>>>(pooled, Wq, bq, query, Dc, Dc, 1.0f,
                                          scalars, 0);
  // base_scores = query @ keys^T * inv_sqrt_d
  const float inv_sqrt_d = 0.022097086912079608f;  // 1/sqrt(2048)
  k_gemm_f32<<<Nc / 16, 128, 0, stream>>>(query, keys, nullptr, bscores, Nc, Dc,
                                          inv_sqrt_d, scalars, 0);

  float rw = 1.0f;
  for (int r = 0; r < ROUNDSc; ++r) {
    k_round_gate<<<1, 1024, 0, stream>>>(prio, valid, wages, effact, scalars, rw);
    k_softmax<<<Bc, 256, 0, stream>>>(bscores, effact, valid, awf, awb);
    k_gemm_out<<<256, 128, 0, stream>>>(awb, vt, part);           // bf16 WMMA
    k_reduce_part<<<(Bc * Dc) / 256, 256, 0, stream>>>(part, acc, scalars);
    k_colmean<<<Nc / 256, 256, 0, stream>>>(awf, wages, scalars);
    rw *= 0.9f;
  }

  // out = (acc / max(wsum,1e-8)) @ Wc^T + bc, zeroed if wsum<=0
  k_gemm_f32<<<Dc / 16, 128, 0, stream>>>(acc, Wc, bc, out, Dc, Dc, 1.0f,
                                          scalars, 1);
}
